// egats_85263690760755
// MI455X (gfx1250) — matrix operand order repaired and checked by reading, not compiled, not run on hardware
//
#include <hip/hip_runtime.h>
#include <math.h>

// ---------------------------------------------------------------------------
// Problem constants (from reference)
// ---------------------------------------------------------------------------
#define NN 4096      // nodes
#define EE 131072    // edges
#define LL 16        // EGAT layers
#define DD 64        // feature dim
#define DM 1024      // model dim (L*D)
#define HEADS 4
#define DH 256       // DM / HEADS
#define KTOP 16      // SortPooling k

// ---------------------------------------------------------------------------
// WMMA types (gfx1250, wave32)
// ---------------------------------------------------------------------------
typedef __attribute__((ext_vector_type(16))) __bf16 v16bf;
typedef __attribute__((ext_vector_type(8)))  __bf16 v8bf;
typedef __attribute__((ext_vector_type(8)))  float  v8f;

// epilogue modes
#define EP_NONE 0
#define EP_RELU 1
#define EP_GELU 2
#define EP_ATTN 3   // v = acc*scale; diag -> -5e4 ; col>row -> -1e9

// ---------------------------------------------------------------------------
// Generic bf16 WMMA GEMM:  C[M x Nc] = act( scale * A[M x K] @ B + bias )
//   A: f32 row-major, leading dim lda
//   B: f32; TB==0: K x Nc (ldb); TB==1: Nc x K (ldb)
//   Requires M%128==0, Nc%64==0, K%32==0 (true for every call site here).
// Block = 256 threads = 8 waves; macro tile 128(M) x 64(N); each wave owns a
// 32x32 C block = 2 M-tiles x 2 N-tiles -> 4 WMMAs per K-step. K step = 32.
// ---------------------------------------------------------------------------
template <int TB, int MODE>
__global__ __launch_bounds__(256) void gemm_bf16_wmma(
    const float* __restrict__ A, int lda,
    const float* __restrict__ B, int ldb,
    float* __restrict__ C, int ldc,
    int Kd, const float* __restrict__ bias, float scale)
{
    __shared__ __bf16 As[128][40];  // [m][k], pad to 40 halves (80B rows, 16B-aligned)
    __shared__ __bf16 Bs[64][40];   // [n][k] (always stored K-transposed)

    const int tid   = threadIdx.x;
    const int lane  = tid & 31;
    const int wave  = tid >> 5;        // 0..7
    const int wm    = wave >> 1;       // 0..3  -> rows wm*32
    const int wn    = wave & 1;        // 0..1  -> cols wn*32
    const int mBase = blockIdx.x * 128;
    const int nBase = blockIdx.y * 64;

    const int lm    = lane & 15;
    const int khalf = lane >> 4;       // 0 or 1

    v8f acc00 = {}, acc01 = {}, acc10 = {}, acc11 = {};

    for (int k0 = 0; k0 < Kd; k0 += 32) {
        // ---- stage A tile: 128 rows x 32 k, 16 floats/thread (4x b128 loads)
        {
            const int r = tid >> 1;
            const int c = (tid & 1) * 16;
            const float4* ap = (const float4*)(A + (size_t)(mBase + r) * lda + (k0 + c));
            float4 x0 = ap[0], x1 = ap[1], x2 = ap[2], x3 = ap[3];
            __bf16* s = &As[r][c];
            s[0]=(__bf16)x0.x;  s[1]=(__bf16)x0.y;  s[2]=(__bf16)x0.z;  s[3]=(__bf16)x0.w;
            s[4]=(__bf16)x1.x;  s[5]=(__bf16)x1.y;  s[6]=(__bf16)x1.z;  s[7]=(__bf16)x1.w;
            s[8]=(__bf16)x2.x;  s[9]=(__bf16)x2.y;  s[10]=(__bf16)x2.z; s[11]=(__bf16)x2.w;
            s[12]=(__bf16)x3.x; s[13]=(__bf16)x3.y; s[14]=(__bf16)x3.z; s[15]=(__bf16)x3.w;
            if (k0 + 32 < Kd)
                __builtin_prefetch(A + (size_t)(mBase + r) * lda + (k0 + 32 + c), 0, 3);
        }
        // ---- stage B tile into [n][k]
        if (TB) {
            const int n = tid >> 2;
            const int c = (tid & 3) * 8;
            const float4* bp = (const float4*)(B + (size_t)(nBase + n) * ldb + (k0 + c));
            float4 x0 = bp[0], x1 = bp[1];
            Bs[n][c+0]=(__bf16)x0.x; Bs[n][c+1]=(__bf16)x0.y;
            Bs[n][c+2]=(__bf16)x0.z; Bs[n][c+3]=(__bf16)x0.w;
            Bs[n][c+4]=(__bf16)x1.x; Bs[n][c+5]=(__bf16)x1.y;
            Bs[n][c+6]=(__bf16)x1.z; Bs[n][c+7]=(__bf16)x1.w;
            if (k0 + 32 < Kd)
                __builtin_prefetch(B + (size_t)(nBase + n) * ldb + (k0 + 32 + c), 0, 3);
        } else {
            const int kr = tid >> 3;          // 0..31
            const int nc = (tid & 7) * 8;     // 0..56
            const float4* bp = (const float4*)(B + (size_t)(k0 + kr) * ldb + (nBase + nc));
            float4 x0 = bp[0], x1 = bp[1];
            Bs[nc+0][kr]=(__bf16)x0.x; Bs[nc+1][kr]=(__bf16)x0.y;
            Bs[nc+2][kr]=(__bf16)x0.z; Bs[nc+3][kr]=(__bf16)x0.w;
            Bs[nc+4][kr]=(__bf16)x1.x; Bs[nc+5][kr]=(__bf16)x1.y;
            Bs[nc+6][kr]=(__bf16)x1.z; Bs[nc+7][kr]=(__bf16)x1.w;
            if (k0 + 32 < Kd)
                __builtin_prefetch(B + (size_t)(k0 + 32 + kr) * ldb + (nBase + nc), 0, 3);
        }
        __syncthreads();

        // ---- A fragments (lane half 0 -> K{0..7,16..23}, half 1 -> K{8..15,24..31})
        v16bf a0, a1;
        {
            const __bf16* p0 = &As[wm * 32 + lm][khalf * 8];
            v8bf l0 = *(const v8bf*)(p0);
            v8bf h0 = *(const v8bf*)(p0 + 16);
            const __bf16* p1 = &As[wm * 32 + 16 + lm][khalf * 8];
            v8bf l1 = *(const v8bf*)(p1);
            v8bf h1 = *(const v8bf*)(p1 + 16);
            #pragma unroll
            for (int j = 0; j < 8; ++j) {
                a0[j] = l0[j]; a0[8 + j] = h0[j];
                a1[j] = l1[j]; a1[8 + j] = h1[j];
            }
        }
        // ---- B fragments (lane half 0 -> K{0..7,16..23}, half 1 -> K{8..15,24..31})
        v16bf b0, b1;
        {
            const __bf16* p0 = &Bs[wn * 32 + lm][khalf * 8];
            v8bf l0 = *(const v8bf*)(p0);
            v8bf h0 = *(const v8bf*)(p0 + 16);
            const __bf16* p1 = &Bs[wn * 32 + 16 + lm][khalf * 8];
            v8bf l1 = *(const v8bf*)(p1);
            v8bf h1 = *(const v8bf*)(p1 + 16);
            #pragma unroll
            for (int j = 0; j < 8; ++j) {
                b0[j] = l0[j]; b0[8 + j] = h0[j];
                b1[j] = l1[j]; b1[8 + j] = h1[j];
            }
        }
        acc00 = __builtin_amdgcn_wmma_f32_16x16x32_bf16(false, a0, false, b0, (short)0, acc00, false, false);
        acc01 = __builtin_amdgcn_wmma_f32_16x16x32_bf16(false, a0, false, b1, (short)0, acc01, false, false);
        acc10 = __builtin_amdgcn_wmma_f32_16x16x32_bf16(false, a1, false, b0, (short)0, acc10, false, false);
        acc11 = __builtin_amdgcn_wmma_f32_16x16x32_bf16(false, a1, false, b1, (short)0, acc11, false, false);
        __syncthreads();
    }

    // ---- epilogue.  C/D layout: VGPR r -> M = r + (lane>=16 ? 8 : 0), N = lane&15
    const int row0 = mBase + wm * 32 + (khalf ? 8 : 0);
    const int col0 = nBase + wn * 32 + lm;
    const int col1 = col0 + 16;
    const float b0v = bias ? bias[col0] : 0.f;
    const float b1v = bias ? bias[col1] : 0.f;

    #pragma unroll
    for (int r = 0; r < 8; ++r) {
        const int gr0 = row0 + r;
        const int gr1 = gr0 + 16;
        float v00 = acc00[r] * scale + b0v;
        float v01 = acc01[r] * scale + b1v;
        float v10 = acc10[r] * scale + b0v;
        float v11 = acc11[r] * scale + b1v;
        if (MODE == EP_RELU) {
            v00 = fmaxf(v00, 0.f); v01 = fmaxf(v01, 0.f);
            v10 = fmaxf(v10, 0.f); v11 = fmaxf(v11, 0.f);
        } else if (MODE == EP_GELU) {
            v00 = 0.5f * v00 * (1.f + erff(v00 * 0.70710678118f));
            v01 = 0.5f * v01 * (1.f + erff(v01 * 0.70710678118f));
            v10 = 0.5f * v10 * (1.f + erff(v10 * 0.70710678118f));
            v11 = 0.5f * v11 * (1.f + erff(v11 * 0.70710678118f));
        } else if (MODE == EP_ATTN) {
            if (gr0 == col0) v00 = -5e4f; else if (gr0 < col0) v00 = -1e9f;
            if (gr0 == col1) v01 = -5e4f; else if (gr0 < col1) v01 = -1e9f;
            if (gr1 == col0) v10 = -5e4f; else if (gr1 < col0) v10 = -1e9f;
            if (gr1 == col1) v11 = -5e4f; else if (gr1 < col1) v11 = -1e9f;
        }
        C[(size_t)gr0 * ldc + col0] = v00;
        C[(size_t)gr0 * ldc + col1] = v01;
        C[(size_t)gr1 * ldc + col0] = v10;
        C[(size_t)gr1 * ldc + col1] = v11;
    }
}

// ---------------------------------------------------------------------------
// Elementwise / scatter / reduction kernels
// ---------------------------------------------------------------------------
__global__ void k_fill(float* p, float v, int n) {
    int i = blockIdx.x * 256 + threadIdx.x;
    if (i < n) p[i] = v;
}

__global__ void k_embed_nodes(const int* __restrict__ tok, const float* __restrict__ emb,
                              float* __restrict__ h, int n) {
    int i = blockIdx.x * 256 + threadIdx.x;
    if (i >= n) return;
    int node = i >> 6, d = i & 63;
    h[i] = fmaxf(emb[tok[node] * DD + d], 0.f);
}

__global__ void k_embed_edges(const int* __restrict__ tok, const float* __restrict__ emb,
                              float* __restrict__ e, int n) {
    int i = blockIdx.x * 256 + threadIdx.x;
    if (i >= n) return;
    int ed = i >> 6, d = i & 63;
    e[i] = emb[tok[ed] * DD + d];
}

// f_out (in-place on eWf) += hWni[src] + hWnj[dst] + bias
__global__ void k_edge_update(float* __restrict__ f,
                              const float* __restrict__ hni, const float* __restrict__ hnj,
                              const int* __restrict__ src, const int* __restrict__ dst,
                              const float* __restrict__ b, int n) {
    int i = blockIdx.x * 256 + threadIdx.x;
    if (i >= n) return;
    int ed = i >> 6, d = i & 63;
    f[i] = f[i] + hni[(size_t)src[ed] * DD + d] + hnj[(size_t)dst[ed] * DD + d] + b[d];
}

__global__ void k_logits(const float* __restrict__ f, const float* __restrict__ a,
                         float* __restrict__ lg, int ne) {
    int ed = blockIdx.x * 256 + threadIdx.x;
    if (ed >= ne) return;
    const float* r = f + (size_t)ed * DD;
    float acc = 0.f;
    #pragma unroll
    for (int d = 0; d < DD; ++d) {
        float x = r[d];
        x = x > 0.f ? x : 0.2f * x;      // leaky_relu(0.2)
        acc += x * a[d];
    }
    lg[ed] = acc;
}

__device__ inline void atomicMaxF(float* addr, float val) {
    if (val >= 0.f) atomicMax((int*)addr, __float_as_int(val));
    else            atomicMin((unsigned int*)addr, (unsigned int)__float_as_int(val));
}

__global__ void k_segmax(const float* __restrict__ lg, const int* __restrict__ dst,
                         float* __restrict__ m, int ne) {
    int ed = blockIdx.x * 256 + threadIdx.x;
    if (ed < ne) atomicMaxF(&m[dst[ed]], lg[ed]);
}

__global__ void k_expsum(const float* __restrict__ lg, const int* __restrict__ dst,
                         const float* __restrict__ m, float* __restrict__ ex,
                         float* __restrict__ z, int ne) {
    int ed = blockIdx.x * 256 + threadIdx.x;
    if (ed >= ne) return;
    float e = expf(lg[ed] - m[dst[ed]]);
    ex[ed] = e;
    atomicAdd(&z[dst[ed]], e);
}

__global__ void k_aggregate(const float* __restrict__ hwn, const int* __restrict__ src,
                            const int* __restrict__ dst, const float* __restrict__ ex,
                            const float* __restrict__ z, float* __restrict__ hn, int n) {
    int i = blockIdx.x * 256 + threadIdx.x;
    if (i >= n) return;
    int ed = i >> 6, d = i & 63;
    float alpha = ex[ed] / z[dst[ed]];
    atomicAdd(&hn[(size_t)dst[ed] * DD + d], hwn[(size_t)src[ed] * DD + d] * alpha);
}

// relu in place; also scatter into hs[:, l*64 : (l+1)*64]
__global__ void k_relu_hs(float* __restrict__ hn, float* __restrict__ hs, int l, int n) {
    int i = blockIdx.x * 256 + threadIdx.x;
    if (i >= n) return;
    int node = i >> 6, d = i & 63;
    float r = fmaxf(hn[i], 0.f);
    hn[i] = r;
    hs[(size_t)node * DM + l * DD + d] = r;
}

// per (node, head): l2-normalize the dh=256 slice of qk -> kn
__global__ void k_knorm(const float* __restrict__ qk, float* __restrict__ kn, int n) {
    int i = blockIdx.x * 256 + threadIdx.x;
    if (i >= n) return;
    const float* p = qk + (size_t)(i >> 2) * DM + (i & 3) * DH;
    float*       o = kn + (size_t)(i >> 2) * DM + (i & 3) * DH;
    float ss = 0.f;
    for (int j = 0; j < DH; ++j) { float v = p[j]; ss += v * v; }
    float inv = 1.f / fmaxf(sqrtf(ss), 1e-12f);
    for (int j = 0; j < DH; ++j) o[j] = p[j] * inv;
}

// row softmax over n columns (one block per row)
__global__ void k_softmax_row(float* __restrict__ S, int n) {
    __shared__ float red[256];
    float* row = S + (size_t)blockIdx.x * n;
    float m = -__builtin_inff();
    for (int j = threadIdx.x; j < n; j += 256) m = fmaxf(m, row[j]);
    red[threadIdx.x] = m; __syncthreads();
    for (int s = 128; s > 0; s >>= 1) {
        if (threadIdx.x < s) red[threadIdx.x] = fmaxf(red[threadIdx.x], red[threadIdx.x + s]);
        __syncthreads();
    }
    m = red[0]; __syncthreads();
    float sum = 0.f;
    for (int j = threadIdx.x; j < n; j += 256) { float e = expf(row[j] - m); row[j] = e; sum += e; }
    red[threadIdx.x] = sum; __syncthreads();
    for (int s = 128; s > 0; s >>= 1) {
        if (threadIdx.x < s) red[threadIdx.x] += red[threadIdx.x + s];
        __syncthreads();
    }
    float inv = 1.f / red[0];
    for (int j = threadIdx.x; j < n; j += 256) row[j] *= inv;
}

// per-node ascending sort of the 64 features; key = max feature
__global__ void k_sortrows(const float* __restrict__ H, float* __restrict__ S,
                           float* __restrict__ keys, int n) {
    int i = blockIdx.x * 256 + threadIdx.x;
    if (i >= n) return;
    float a[DD];
    for (int d = 0; d < DD; ++d) a[d] = H[(size_t)i * DD + d];
    for (int p = 1; p < DD; ++p) {
        float key = a[p]; int q = p - 1;
        while (q >= 0 && a[q] > key) { a[q + 1] = a[q]; --q; }
        a[q + 1] = key;
    }
    for (int d = 0; d < DD; ++d) S[(size_t)i * DD + d] = a[d];
    keys[i] = a[DD - 1];
}

// top-16 argmax (stable: lower index wins ties), single block
__global__ void k_topk(const float* __restrict__ keys, int n, int* __restrict__ topk) {
    __shared__ float kv[NN];
    __shared__ float bv[256];
    __shared__ int   bi[256];
    for (int j = threadIdx.x; j < n; j += 256) kv[j] = keys[j];
    __syncthreads();
    for (int t = 0; t < KTOP; ++t) {
        float best = -__builtin_inff(); int bidx = 0x7fffffff;
        for (int j = threadIdx.x; j < n; j += 256) {
            float v = kv[j];
            if (v > best || (v == best && j < bidx)) { best = v; bidx = j; }
        }
        bv[threadIdx.x] = best; bi[threadIdx.x] = bidx; __syncthreads();
        for (int s = 128; s > 0; s >>= 1) {
            if (threadIdx.x < s) {
                if (bv[threadIdx.x + s] > bv[threadIdx.x] ||
                    (bv[threadIdx.x + s] == bv[threadIdx.x] && bi[threadIdx.x + s] < bi[threadIdx.x])) {
                    bv[threadIdx.x] = bv[threadIdx.x + s];
                    bi[threadIdx.x] = bi[threadIdx.x + s];
                }
            }
            __syncthreads();
        }
        if (threadIdx.x == 0) { topk[t] = bi[0]; kv[bi[0]] = -__builtin_inff(); }
        __syncthreads();
    }
}

// pooled gather + 3-layer classifier head, single block
__global__ void k_final(const float* __restrict__ Ssorted, const int* __restrict__ topk,
                        const float* __restrict__ Wl, const float* __restrict__ bl,
                        const float* __restrict__ Wl1, const float* __restrict__ bl1,
                        const float* __restrict__ Wc, const float* __restrict__ bc,
                        float* __restrict__ out) {
    __shared__ float pooled[KTOP * DD];
    __shared__ float s1[DD];
    __shared__ float s2[DD];
    int t = threadIdx.x;
    for (int j = t; j < KTOP * DD; j += 256) {
        int ki = j >> 6, d = j & 63;
        pooled[j] = Ssorted[(size_t)topk[ki] * DD + d];
    }
    __syncthreads();
    if (t < DD) {
        float acc = bl[t];
        for (int j = 0; j < KTOP * DD; ++j) acc += pooled[j] * Wl[j * DD + t];
        s1[t] = fmaxf(acc, 0.f);
    }
    __syncthreads();
    if (t < DD) {
        float acc = bl1[t];
        for (int j = 0; j < DD; ++j) acc += s1[j] * Wl1[j * DD + t];
        s2[t] = fmaxf(acc, 0.f);
    }
    __syncthreads();
    if (t < 2) {
        float acc = bc[t];
        for (int j = 0; j < DD; ++j) acc += s2[j] * Wc[j * 2 + t];
        out[t] = acc;
    }
}

// ---------------------------------------------------------------------------
// Host side
// ---------------------------------------------------------------------------
static inline void launch_gemm(const float* A, int lda, const float* B, int ldb,
                               float* C, int ldc, int M, int Nc, int Kd,
                               const float* bias, int mode, float scale, int transB,
                               hipStream_t s) {
    dim3 g(M / 128, Nc / 64);
    if (transB) {
        gemm_bf16_wmma<1, EP_ATTN><<<g, 256, 0, s>>>(A, lda, B, ldb, C, ldc, Kd, bias, scale);
    } else if (mode == EP_RELU) {
        gemm_bf16_wmma<0, EP_RELU><<<g, 256, 0, s>>>(A, lda, B, ldb, C, ldc, Kd, bias, scale);
    } else if (mode == EP_GELU) {
        gemm_bf16_wmma<0, EP_GELU><<<g, 256, 0, s>>>(A, lda, B, ldb, C, ldc, Kd, bias, scale);
    } else {
        gemm_bf16_wmma<0, EP_NONE><<<g, 256, 0, s>>>(A, lda, B, ldb, C, ldc, Kd, bias, scale);
    }
}

#define GRID1(n) dim3((unsigned)(((n) + 255) / 256)), dim3(256)

extern "C" void kernel_launch(void* const* d_in, const int* in_sizes, int n_in,
                              void* d_out, int out_size, void* d_ws, size_t ws_size,
                              hipStream_t stream) {
    const int*   h_tokens    = (const int*)d_in[0];
    const int*   e_tokens    = (const int*)d_in[1];
    const int*   src         = (const int*)d_in[2];
    const int*   dst         = (const int*)d_in[3];
    const float* token_emb   = (const float*)d_in[4];
    const float* e_token_emb = (const float*)d_in[5];
    const float* Wn   = (const float*)d_in[6];
    const float* Wni  = (const float*)d_in[7];
    const float* Wnj  = (const float*)d_in[8];
    const float* Wf   = (const float*)d_in[9];
    const float* attn_p = (const float*)d_in[10];
    const float* bias_p = (const float*)d_in[11];
    const float* Wqk  = (const float*)d_in[12];
    const float* Wv   = (const float*)d_in[13];
    const float* Wout = (const float*)d_in[14];
    const float* bout = (const float*)d_in[15];
    const float* Wi1  = (const float*)d_in[16];
    const float* bi1  = (const float*)d_in[17];
    const float* Wi2  = (const float*)d_in[18];
    const float* bi2  = (const float*)d_in[19];
    const float* Wf1  = (const float*)d_in[20];
    const float* bf1  = (const float*)d_in[21];
    const float* Wl   = (const float*)d_in[22];
    const float* bl   = (const float*)d_in[23];
    const float* Wl1  = (const float*)d_in[24];
    const float* bl1  = (const float*)d_in[25];
    const float* Wc   = (const float*)d_in[26];
    const float* bc   = (const float*)d_in[27];
    float* out = (float*)d_out;

    // ---- workspace bump allocator (~245 MB total)
    float* w = (float*)d_ws;
    auto alloc = [&](size_t n) { float* p = w; w += n; return p; };
    float* h_a   = alloc((size_t)NN * DD);
    float* h_b   = alloc((size_t)NN * DD);
    float* hWniB = alloc((size_t)NN * DD);
    float* hWnjB = alloc((size_t)NN * DD);
    float* hWnB  = alloc((size_t)NN * DD);
    float* e_a   = alloc((size_t)EE * DD);
    float* e_b   = alloc((size_t)EE * DD);
    float* lg    = alloc((size_t)EE);
    float* exv   = alloc((size_t)EE);
    float* segm  = alloc((size_t)NN);
    float* segz  = alloc((size_t)NN);
    float* hs    = alloc((size_t)NN * DM);
    float* qk    = alloc((size_t)NN * DM);
    float* vv    = alloc((size_t)NN * DM);
    float* kn    = alloc((size_t)NN * DM);
    float* obuf  = alloc((size_t)NN * DM);
    float* o2    = alloc((size_t)NN * DM);
    float* t1    = alloc((size_t)NN * DD);
    float* t2    = alloc((size_t)NN * DM);
    float* hfin  = alloc((size_t)NN * DD);
    float* hsort = alloc((size_t)NN * DD);
    float* keys  = alloc((size_t)NN);
    int*   tk    = (int*)alloc(32);
    float* scores = alloc((size_t)NN * NN);
    (void)ws_size; (void)in_sizes; (void)n_in; (void)out_size;

    // ---- embeddings
    k_embed_nodes<<<GRID1(NN * DD), 0, stream>>>(h_tokens, token_emb, h_a, NN * DD);
    k_embed_edges<<<GRID1(EE * DD), 0, stream>>>(e_tokens, e_token_emb, e_a, EE * DD);

    // ---- 16 EGAT layers
    float* hcur = h_a; float* hnew = h_b;
    float* ecur = e_a; float* enew = e_b;
    for (int l = 0; l < LL; ++l) {
        const float* wni = Wni + (size_t)l * DD * DD;
        const float* wnj = Wnj + (size_t)l * DD * DD;
        const float* wn  = Wn  + (size_t)l * DD * DD;
        const float* wf  = Wf  + (size_t)l * DD * DD;
        const float* ap  = attn_p + (size_t)l * DD;
        const float* bp  = bias_p + (size_t)l * DD;

        // node-side GEMMs (h[src]@W == (h@W)[src])
        launch_gemm(hcur, DD, wni, DD, hWniB, DD, NN, DD, DD, nullptr, EP_NONE, 1.f, 0, stream);
        launch_gemm(hcur, DD, wnj, DD, hWnjB, DD, NN, DD, DD, nullptr, EP_NONE, 1.f, 0, stream);
        launch_gemm(hcur, DD, wn,  DD, hWnB,  DD, NN, DD, DD, nullptr, EP_NONE, 1.f, 0, stream);
        // edge GEMM: enew = ecur @ Wf
        launch_gemm(ecur, DD, wf, DD, enew, DD, EE, DD, DD, nullptr, EP_NONE, 1.f, 0, stream);
        // f_out = enew + hWni[src] + hWnj[dst] + bias  (in place)
        k_edge_update<<<GRID1(EE * DD), 0, stream>>>(enew, hWniB, hWnjB, src, dst, bp, EE * DD);
        // logits = leaky_relu(f_out) @ a
        k_logits<<<GRID1(EE), 0, stream>>>(enew, ap, lg, EE);
        // edge softmax over dst
        k_fill<<<GRID1(NN), 0, stream>>>(segm, -__builtin_inff(), NN);
        k_fill<<<GRID1(NN), 0, stream>>>(segz, 0.f, NN);
        k_segmax<<<GRID1(EE), 0, stream>>>(lg, dst, segm, EE);
        k_expsum<<<GRID1(EE), 0, stream>>>(lg, dst, segm, exv, segz, EE);
        // aggregate
        k_fill<<<GRID1(NN * DD), 0, stream>>>(hnew, 0.f, NN * DD);
        k_aggregate<<<GRID1(EE * DD), 0, stream>>>(hWnB, src, dst, exv, segz, hnew, EE * DD);
        // relu + write layer slice of hs
        k_relu_hs<<<GRID1(NN * DD), 0, stream>>>(hnew, hs, l, NN * DD);

        { float* t = hcur; hcur = hnew; hnew = t; }
        { float* t = ecur; ecur = enew; enew = t; }
    }

    // ---- attention projections
    launch_gemm(hs, DM, Wqk, DM, qk, DM, NN, DM, DM, nullptr, EP_NONE, 1.f, 0, stream);
    launch_gemm(hs, DM, Wv,  DM, vv, DM, NN, DM, DM, nullptr, EP_NONE, 1.f, 0, stream);
    k_knorm<<<GRID1(NN * HEADS), 0, stream>>>(qk, kn, NN * HEADS);

    const float invSqrtDh = 0.0625f;   // 256^-0.5
    for (int hd = 0; hd < HEADS; ++hd) {
        const float* qh = qk + (size_t)hd * DH;
        const float* kh = kn + (size_t)hd * DH;
        const float* vh = vv + (size_t)hd * DH;
        float*       oh = obuf + (size_t)hd * DH;
        // scores = scale * qh @ kh^T  (+ diag / causal mask)
        launch_gemm(qh, DM, kh, DM, scores, NN, NN, NN, DH, nullptr, EP_ATTN, invSqrtDh, 1, stream);
        k_softmax_row<<<dim3(NN), dim3(256), 0, stream>>>(scores, NN);
        // oh = scores @ vh
        launch_gemm(scores, NN, vh, DM, oh, DM, NN, DH, NN, nullptr, EP_NONE, 1.f, 0, stream);
    }

    // ---- output projection + MLPs
    launch_gemm(obuf, DM, Wout, DM, o2, DM, NN, DM, DM, bout, EP_NONE, 1.f, 0, stream);
    launch_gemm(o2, DM, Wi1, DD, t1, DD, NN, DD, DM, bi1, EP_GELU, 1.f, 0, stream);
    launch_gemm(t1, DD, Wi2, DM, t2, DM, NN, DM, DD, bi2, EP_GELU, 1.f, 0, stream);
    launch_gemm(t2, DM, Wf1, DD, hfin, DD, NN, DD, DM, bf1, EP_RELU, 1.f, 0, stream);

    // ---- SortPooling + head
    k_sortrows<<<GRID1(NN), 0, stream>>>(hfin, hsort, keys, NN);
    k_topk<<<dim3(1), dim3(256), 0, stream>>>(keys, NN, tk);
    k_final<<<dim3(1), dim3(256), 0, stream>>>(hsort, tk, Wl, bl, Wl1, bl1, Wc, bc, out);
}
